// GlobalTransformer_68143951118508
// MI455X (gfx1250) — matrix operand order repaired and checked by reading, not compiled, not run on hardware
//
#include <hip/hip_runtime.h>
#include <hip/hip_bf16.h>

typedef __attribute__((ext_vector_type(16))) __bf16 v16bf;
typedef __attribute__((ext_vector_type(8)))  float  v8f;
typedef __attribute__((ext_vector_type(4)))  unsigned int uint4v;
typedef __attribute__((ext_vector_type(4)))  float  float4v;

#define B_SZ   8192
#define F_IN   2048
#define NVERT  2334           // 778*3
#define K2_RAW (F_IN + NVERT) // 4382
#define K2_PAD 4384           // padded to multiple of 32 (zero-filled -> exact)
#define NHEAD  12

__device__ __forceinline__ unsigned short f2bf(float x) {
  unsigned int u = __float_as_uint(x);
  u += 0x7FFFu + ((u >> 16) & 1u);   // round-to-nearest-even
  return (unsigned short)(u >> 16);
}
__device__ __forceinline__ float bf2f(unsigned short h) {
  return __uint_as_float(((unsigned int)h) << 16);
}

// ---------------- conversion / packing kernels ----------------
// n8 = n/8 (element counts are multiples of 8)
__global__ void k_f32_to_bf16_v8(const float* __restrict__ src,
                                 unsigned short* __restrict__ dst, int n8) {
  int i = blockIdx.x * blockDim.x + threadIdx.x;
  if (i >= n8) return;
  const float4v f0 = *(const float4v*)(src + (size_t)i * 8);
  const float4v f1 = *(const float4v*)(src + (size_t)i * 8 + 4);
  unsigned short o[8];
  o[0]=f2bf(f0.x); o[1]=f2bf(f0.y); o[2]=f2bf(f0.z); o[3]=f2bf(f0.w);
  o[4]=f2bf(f1.x); o[5]=f2bf(f1.y); o[6]=f2bf(f1.z); o[7]=f2bf(f1.w);
  *(uint4v*)(dst + (size_t)i * 8) = *(const uint4v*)o;
}

__global__ void k_convert_w1(const float* __restrict__ w1,
                             unsigned short* __restrict__ dst) {
  int i = blockIdx.x * blockDim.x + threadIdx.x;
  if (i >= F_IN * K2_PAD) return;
  int r = i / K2_PAD, c = i - r * K2_PAD;
  dst[i] = (c < K2_RAW) ? f2bf(w1[(size_t)r * K2_RAW + c]) : (unsigned short)0;
}

__global__ void k_pack_verts(const float* __restrict__ verts,
                             unsigned short* __restrict__ x) {
  int i = blockIdx.x * blockDim.x + threadIdx.x;
  if (i >= B_SZ * (NVERT + 2)) return;
  int r = i / (NVERT + 2), c = i - r * (NVERT + 2);
  if (c < NVERT) x[(size_t)r * K2_PAD + c] = f2bf(verts[(size_t)r * NVERT + c]);
  else           x[(size_t)r * K2_PAD + K2_RAW + (c - NVERT)] = 0; // zero pad cols
}

// ---------------- bf16 WMMA GEMM: C = relu(A @ W^T + bias), bf16 out -------
// A: [M x K] bf16 row-major (lda), W: [N x K] bf16 row-major (ldb),
// C: [M x N] bf16 (ldc). M,N multiples of 128; K multiple of 32.
// Structurally double-buffered LDS (2-phase loop): one barrier per K-step,
// constant LDS addresses, 2x-unrolled steady state.
#define BM 128
#define BN 128
#define BK 32
#define LSTR 40   // LDS row stride in ushorts (80B) -> conflict-free b128 reads

__global__ __launch_bounds__(256)
void k_gemm_relu_bf16(const unsigned short* __restrict__ A, int lda,
                      const unsigned short* __restrict__ Bw, int ldb,
                      const float* __restrict__ bias,
                      unsigned short* __restrict__ C, int ldc, int K) {
  __shared__ unsigned short Asm0[BM * LSTR];
  __shared__ unsigned short Bsm0[BN * LSTR];
  __shared__ unsigned short Asm1[BM * LSTR];
  __shared__ unsigned short Bsm1[BN * LSTR];

  const int tid  = threadIdx.x;
  const int lane = tid & 31;
  const int wave = tid >> 5;       // 8 waves (wave32)
  const int wm   = wave >> 2;      // 0..1 -> 64-row slab
  const int wn   = wave & 3;       // 0..3 -> 32-col slab
  const int bm0  = blockIdx.x * BM;
  const int bn0  = blockIdx.y * BN;

  const int lrow  = tid >> 1;            // 0..127: tile row this thread loads
  const int lhalf = (tid & 1) << 4;      // 0 or 16 ushorts (half a 32-wide row)
  const int soff  = lrow * LSTR + lhalf; // LDS store offset (ushorts)

  v8f acc[4][2];
#pragma unroll
  for (int mi = 0; mi < 4; ++mi)
#pragma unroll
    for (int ni = 0; ni < 2; ++ni)
      acc[mi][ni] = (v8f){0.f,0.f,0.f,0.f,0.f,0.f,0.f,0.f};

  const unsigned short* Ag = A  + (size_t)(bm0 + lrow) * lda + lhalf;
  const unsigned short* Bg = Bw + (size_t)(bn0 + lrow) * ldb + lhalf;

  union FragU { v16bf v; uint4v u[2]; };

  // A fragment (16x32 bf16): lanes 0-15 row m, K {0..7, 16..23};
  //                          lanes 16-31 row m, K {8..15, 24..31}
  // B fragment (32x16): lane holds col n = lane%16, same K order as A:
  // lanes 0-15 K {0..7, 16..23}; lanes 16-31 K {8..15, 24..31}  (B[k][n] = W[n][k])
  const int arow = lane & 15;
  const int akb  = (lane >> 4) << 3;   // 0 or 8
  const int bkb  = (lane >> 4) << 3;   // 0 or 8

  auto compute_step = [&](const unsigned short* As, const unsigned short* Bs) {
    FragU afrag[4], bfrag[2];
#pragma unroll
    for (int mi = 0; mi < 4; ++mi) {
      const unsigned short* p = &As[(wm * 64 + mi * 16 + arow) * LSTR + akb];
      afrag[mi].u[0] = *(const uint4v*)(p);
      afrag[mi].u[1] = *(const uint4v*)(p + 16);
    }
#pragma unroll
    for (int ni = 0; ni < 2; ++ni) {
      const unsigned short* p = &Bs[(wn * 32 + ni * 16 + (lane & 15)) * LSTR + bkb];
      bfrag[ni].u[0] = *(const uint4v*)(p);
      bfrag[ni].u[1] = *(const uint4v*)(p + 16);
    }
#pragma unroll
    for (int mi = 0; mi < 4; ++mi)
#pragma unroll
      for (int ni = 0; ni < 2; ++ni)
        acc[mi][ni] = __builtin_amdgcn_wmma_f32_16x16x32_bf16(
            false, afrag[mi].v, false, bfrag[ni].v,
            (short)0, acc[mi][ni], false, false);
  };

  // ---- prologue: tile 0 into buffer 0 ----
  {
    uint4v a0 = *(const uint4v*)(Ag);
    uint4v a1 = *(const uint4v*)(Ag + 8);
    uint4v b0 = *(const uint4v*)(Bg);
    uint4v b1 = *(const uint4v*)(Bg + 8);
    *(uint4v*)(&Asm0[soff])     = a0;
    *(uint4v*)(&Asm0[soff + 8]) = a1;
    *(uint4v*)(&Bsm0[soff])     = b0;
    *(uint4v*)(&Bsm0[soff + 8]) = b1;
  }

  int k0 = 0;
  for (;;) {
    // ---- phase 0: compute from buf0, prefetch k0+BK into buf1 ----
    {
      const bool more = (k0 + BK) < K;
      uint4v a0, a1, b0, b1;
      if (more) {
        a0 = *(const uint4v*)(Ag + k0 + BK);
        a1 = *(const uint4v*)(Ag + k0 + BK + 8);
        b0 = *(const uint4v*)(Bg + k0 + BK);
        b1 = *(const uint4v*)(Bg + k0 + BK + 8);
      }
      __syncthreads();
      compute_step(Asm0, Bsm0);
      if (more) {
        *(uint4v*)(&Asm1[soff])     = a0;
        *(uint4v*)(&Asm1[soff + 8]) = a1;
        *(uint4v*)(&Bsm1[soff])     = b0;
        *(uint4v*)(&Bsm1[soff + 8]) = b1;
      }
      k0 += BK;
      if (k0 >= K) break;
    }
    // ---- phase 1: compute from buf1, prefetch k0+BK into buf0 ----
    {
      const bool more = (k0 + BK) < K;
      uint4v a0, a1, b0, b1;
      if (more) {
        a0 = *(const uint4v*)(Ag + k0 + BK);
        a1 = *(const uint4v*)(Ag + k0 + BK + 8);
        b0 = *(const uint4v*)(Bg + k0 + BK);
        b1 = *(const uint4v*)(Bg + k0 + BK + 8);
      }
      __syncthreads();
      compute_step(Asm1, Bsm1);
      if (more) {
        *(uint4v*)(&Asm0[soff])     = a0;
        *(uint4v*)(&Asm0[soff + 8]) = a1;
        *(uint4v*)(&Bsm0[soff])     = b0;
        *(uint4v*)(&Bsm0[soff + 8]) = b1;
      }
      k0 += BK;
      if (k0 >= K) break;
    }
  }

  // Epilogue: D layout -> vgpr r: M = r + 8*(lane/16), N = lane%16
  const int rbase = (lane >> 4) << 3;
#pragma unroll
  for (int mi = 0; mi < 4; ++mi) {
#pragma unroll
    for (int ni = 0; ni < 2; ++ni) {
      const int gn  = bn0 + wn * 32 + ni * 16 + (lane & 15);
      const float bv = bias[gn];
      const int gm0 = bm0 + wm * 64 + mi * 16 + rbase;
#pragma unroll
      for (int r = 0; r < 8; ++r) {
        float v = acc[mi][ni][r] + bv;
        v = v > 0.f ? v : 0.f;
        C[(size_t)(gm0 + r) * ldc + gn] = f2bf(v);
      }
    }
  }
}

// ---------------- head: t = h @ W2^T + b2, then polar(M) ----------------
// One block handles HROWS batch rows; W2 (96 KB) staged in LDS once.
// Each wave owns one row: wave-local shfl reduction, no block syncs in loop.
#define HROWS 16

__global__ __launch_bounds__(256)
void k_head(const unsigned short* __restrict__ h,
            const float* __restrict__ W2, const float* __restrict__ b2,
            float* __restrict__ params) {
  __shared__ float w2s[NHEAD * F_IN];   // 96 KB
  const int tid = threadIdx.x;
  for (int i = tid; i < NHEAD * F_IN; i += 256) w2s[i] = W2[i];
  __syncthreads();

  const int wave = tid >> 5, lane = tid & 31;
  for (int rr = 0; rr < HROWS; rr += 8) {
    const int b = blockIdx.x * HROWS + rr + wave;
    const unsigned short* hrow = h + (size_t)b * F_IN;
    float acc[NHEAD];
#pragma unroll
    for (int j = 0; j < NHEAD; ++j) acc[j] = 0.f;
    for (int k = lane; k < F_IN; k += 32) {
      const float hv = bf2f(hrow[k]);
#pragma unroll
      for (int j = 0; j < NHEAD; ++j) acc[j] += hv * w2s[j * F_IN + k];
    }
#pragma unroll
    for (int s = 16; s > 0; s >>= 1)
#pragma unroll
      for (int j = 0; j < NHEAD; ++j) acc[j] += __shfl_down(acc[j], s, 32);

    if (lane == 0) {
      float t[NHEAD];
#pragma unroll
      for (int j = 0; j < NHEAD; ++j) t[j] = acc[j] + b2[j];
      float M[9], X[9], Cf[9];
#pragma unroll
      for (int i = 0; i < 9; ++i) { M[i] = t[i]; X[i] = t[i]; }
      // Scaled Newton polar iteration: X <- 0.5*(g*X + (1/g)*X^-T)
      for (int it = 0; it < 10; ++it) {
        Cf[0] = X[4]*X[8] - X[5]*X[7];
        Cf[1] = X[5]*X[6] - X[3]*X[8];
        Cf[2] = X[3]*X[7] - X[4]*X[6];
        Cf[3] = X[2]*X[7] - X[1]*X[8];
        Cf[4] = X[0]*X[8] - X[2]*X[6];
        Cf[5] = X[1]*X[6] - X[0]*X[7];
        Cf[6] = X[1]*X[5] - X[2]*X[4];
        Cf[7] = X[2]*X[3] - X[0]*X[5];
        Cf[8] = X[0]*X[4] - X[1]*X[3];
        float det = X[0]*Cf[0] + X[1]*Cf[1] + X[2]*Cf[2];
        float g = powf(fmaxf(fabsf(det), 1e-12f), -1.0f / 3.0f);
        g = fminf(fmaxf(g, 0.2f), 5.0f);
        const float hg = 0.5f * g;
        const float ig = 0.5f / (g * det);
#pragma unroll
        for (int i = 0; i < 9; ++i) X[i] = hg * X[i] + ig * Cf[i];
      }
      float ssum = 0.f;               // trace(R^T M) = sum of singular values
#pragma unroll
      for (int i = 0; i < 9; ++i) ssum += X[i] * M[i];
      float* p = params + (size_t)b * 12;
#pragma unroll
      for (int i = 0; i < 9; ++i) p[i] = X[i];
      p[9]  = ssum * t[11];           // ssum * scale
      p[10] = t[9];                   // shift x
      p[11] = t[10];                  // shift y
    }
  }
}

// ---------------- final transform: (verts @ R) * k (+shift on xy) -----------
__global__ __launch_bounds__(256)
void k_transform(const float* __restrict__ verts,
                 const float* __restrict__ params,
                 float* __restrict__ out) {
  const int b = blockIdx.y;
  const int v = blockIdx.x * blockDim.x + threadIdx.x;
  if (v >= 778) return;
  const float* p = params + (size_t)b * 12;
  const float R0=p[0],R1=p[1],R2=p[2],R3=p[3],R4=p[4],R5=p[5],R6=p[6],R7=p[7],R8=p[8];
  const float kk=p[9], sx=p[10], sy=p[11];
  const size_t base = ((size_t)b * 778 + v) * 3;
  const float x = verts[base], y = verts[base + 1], z = verts[base + 2];
  out[base]     = (x*R0 + y*R3 + z*R6) * kk + sx;
  out[base + 1] = (x*R1 + y*R4 + z*R7) * kk + sy;
  out[base + 2] = (x*R2 + y*R5 + z*R8) * kk;
}

extern "C" void kernel_launch(void* const* d_in, const int* in_sizes, int n_in,
                              void* d_out, int out_size, void* d_ws, size_t ws_size,
                              hipStream_t stream) {
  (void)in_sizes; (void)n_in; (void)out_size; (void)ws_size;
  const float* verts = (const float*)d_in[0];
  const float* feats = (const float*)d_in[1];
  const float* W_re  = (const float*)d_in[2];
  const float* b_re  = (const float*)d_in[3];
  const float* W1    = (const float*)d_in[4];
  const float* b1    = (const float*)d_in[5];
  const float* W2    = (const float*)d_in[6];
  const float* b2    = (const float*)d_in[7];
  float* out = (float*)d_out;

  char* ws = (char*)d_ws;
  size_t off = 0;
  auto alloc = [&](size_t bytes) -> void* {
    void* p = ws + off;
    off += (bytes + 255) & ~(size_t)255;
    return p;
  };
  unsigned short* featb  = (unsigned short*)alloc((size_t)B_SZ * F_IN   * 2);
  unsigned short* wreb   = (unsigned short*)alloc((size_t)F_IN * F_IN   * 2);
  unsigned short* w1b    = (unsigned short*)alloc((size_t)F_IN * K2_PAD * 2);
  unsigned short* xbuf   = (unsigned short*)alloc((size_t)B_SZ * K2_PAD * 2);
  unsigned short* hbuf   = (unsigned short*)alloc((size_t)B_SZ * F_IN   * 2);
  float*          params = (float*)         alloc((size_t)B_SZ * 12     * 4);

  int n;
  n = (B_SZ * F_IN) / 8;
  k_f32_to_bf16_v8<<<(n + 255) / 256, 256, 0, stream>>>(feats, featb, n);
  n = (F_IN * F_IN) / 8;
  k_f32_to_bf16_v8<<<(n + 255) / 256, 256, 0, stream>>>(W_re, wreb, n);
  n = F_IN * K2_PAD;
  k_convert_w1<<<(n + 255) / 256, 256, 0, stream>>>(W1, w1b);
  n = B_SZ * (NVERT + 2);
  k_pack_verts<<<(n + 255) / 256, 256, 0, stream>>>(verts, xbuf);

  dim3 gg(B_SZ / BM, F_IN / BN);   // 64 x 16
  // GEMM1: f = relu(features @ W_re^T) -> x[:, 2334:4382]
  k_gemm_relu_bf16<<<gg, 256, 0, stream>>>(featb, F_IN, wreb, F_IN, b_re,
                                           xbuf + NVERT, K2_PAD, F_IN);
  // GEMM2: h = relu(x @ W1^T)
  k_gemm_relu_bf16<<<gg, 256, 0, stream>>>(xbuf, K2_PAD, w1b, K2_PAD, b1,
                                           hbuf, F_IN, K2_PAD);

  k_head<<<B_SZ / HROWS, 256, 0, stream>>>(hbuf, W2, b2, params);

  dim3 gt((778 + 255) / 256, B_SZ);
  k_transform<<<gt, 256, 0, stream>>>(verts, params, out);
}
